// GatedAttention_10788957847625
// MI455X (gfx1250) — compile-verified, matrix operand order checked by reading, not run on hardware
//
#include <hip/hip_runtime.h>
#include <stdint.h>

// ---------------- problem constants ----------------
#define T_TOK   4096
#define MDIM    512
#define NLAYERS 2
#define NHEADS  8
#define HDIM    64
#define FFDIM   2048
#define LGATE   128
#define NINNER  256
#define NOUTER  32

// ---------------- CDNA5 WMMA types ----------------
typedef __attribute__((ext_vector_type(16))) __bf16        v16bf;
typedef __attribute__((ext_vector_type(8)))  float         v8f;
typedef __attribute__((ext_vector_type(4)))  unsigned int  u32x4;

union Frag {
    v16bf         v;
    unsigned short u[16];
    u32x4         q[2];
};

__device__ __forceinline__ unsigned short f2bf(float f) {
    unsigned int u = __float_as_uint(f);
    u += 0x7FFFu + ((u >> 16) & 1u);       // round-to-nearest-even
    return (unsigned short)(u >> 16);
}

// =====================================================================
// Generic bf16 WMMA GEMM: C[M,N] = act(A[M,K] @ B[N,K]^T + bias[N])
// One wave -> 16x64 output strip (4 x 16x16 tiles).
// Register double-buffered over K (step 32, ping-pong over 64): fragment
// loads for step k+32 are issued before the WMMAs of step k, so the
// scheduler can overlap global loads with matrix ops instead of emitting
// s_wait_loadcnt 0 in front of every v_wmma.
// Requires K % 64 == 0 and K >= 64 (true for all uses: 512 / 2048).
// =====================================================================
__global__ void __launch_bounds__(256) gemm_bf16_kernel(
    const unsigned short* __restrict__ A, const unsigned short* __restrict__ B,
    const float* __restrict__ bias, float* __restrict__ C,
    unsigned short* __restrict__ Cbf, int M, int N, int K, int act)
{
    const int lane = threadIdx.x & 31;
    const int wave = blockIdx.x * (blockDim.x >> 5) + (threadIdx.x >> 5);
    const int colGroups = N >> 6;              // N / 64
    const int rowTiles  = M >> 4;              // M / 16
    const int rowTile   = wave / colGroups;
    const int cg        = wave - rowTile * colGroups;
    if (rowTile >= rowTiles) return;           // wave-uniform exit: EXEC stays all-1s

    const int m0 = rowTile << 4;
    const int n0 = cg << 6;
    const int kb = (lane < 16) ? 0 : 8;        // 16-bit A/B fragment K-base per half-wave
    const int nr = lane & 15;
    const int mr = m0 + nr;

    const v8f zero = {0.f,0.f,0.f,0.f,0.f,0.f,0.f,0.f};
    v8f acc[4] = {zero, zero, zero, zero};

    const unsigned short* ap    = A + (size_t)mr * K + kb;
    const unsigned short* bbase = B + (size_t)(n0 + nr) * K + kb;

    auto LD = [&](Frag& a, Frag b[4], int k0) {
        a.q[0] = *(const u32x4*)(ap + k0);         // K = k0+kb    .. +7
        a.q[1] = *(const u32x4*)(ap + k0 + 16);    // K = k0+16+kb .. +7
        __builtin_prefetch(ap + k0 + 128, 0, 1);   // global_prefetch_b8
#pragma unroll
        for (int j = 0; j < 4; ++j) {
            const unsigned short* p = bbase + (size_t)j * 16 * K + k0;
            b[j].q[0] = *(const u32x4*)p;
            b[j].q[1] = *(const u32x4*)(p + 16);
        }
    };
    auto MM = [&](Frag& a, Frag b[4]) {
#pragma unroll
        for (int j = 0; j < 4; ++j)
            acc[j] = __builtin_amdgcn_wmma_f32_16x16x32_bf16(
                false, a.v, false, b[j].v, (short)0, acc[j], false, false);
    };

    Frag a0, a1, b0[4], b1[4];
    LD(a0, b0, 0);
    int k0 = 0;
    for (; k0 + 64 < K; k0 += 64) {
        LD(a1, b1, k0 + 32);   // issue next loads ...
        MM(a0, b0);            // ... then consume previous fragments
        LD(a0, b0, k0 + 64);
        MM(a1, b1);
    }
    LD(a1, b1, k0 + 32);       // tail: K-32 fragments
    MM(a0, b0);
    MM(a1, b1);

    const int rofs = (lane >= 16) ? 8 : 0;
#pragma unroll
    for (int j = 0; j < 4; ++j) {
        const int col = n0 + j*16 + nr;
        const float bv = bias ? bias[col] : 0.f;
#pragma unroll
        for (int v = 0; v < 8; ++v) {
            const int row = m0 + v + rofs;
            float val = acc[j][v] + bv;
            if (act == 1) val = fmaxf(val, 0.f);
            const size_t o = (size_t)row * N + col;
            C[o] = val;
            if (Cbf) Cbf[o] = f2bf(val);
        }
    }
}

// =====================================================================
// Flash-style attention: one wave per (head, 16-query tile).
// Streams 32-key blocks: 4 WMMA scores, online softmax, 4 WMMA P@V.
// Q/K in bf16 [head][t][dim]; V in bf16 TRANSPOSED [head][dim][t] so the
// V B-fragments are two b128 loads (issued before the softmax VALU block
// to hide their latency). Output bf16 [t][h*64+d].
// =====================================================================
__global__ void __launch_bounds__(256) attn_kernel(
    const unsigned short* __restrict__ Qh, const unsigned short* __restrict__ Kh,
    const unsigned short* __restrict__ Vt, unsigned short* __restrict__ Obf)
{
    __shared__ unsigned short pst_all[8][16 * 32];   // per-wave P staging (1KB each)
    const int lane = threadIdx.x & 31;
    const int wib  = threadIdx.x >> 5;
    unsigned short* pst = pst_all[wib];

    const int gw    = blockIdx.x * 8 + wib;
    const int numQT = T_TOK / 16;
    const int h     = gw / numQT;
    const int qt    = gw - h * numQT;

    const int kb   = (lane < 16) ? 0 : 8;
    const int nr   = lane & 15;
    const int rofs = (lane >= 16) ? 8 : 0;
    const float scale = 0.125f;                      // 1/sqrt(64)

    // Q fragments: head-dim slices [0,32) and [32,64)
    const unsigned short* qbase = Qh + ((size_t)h * T_TOK + qt*16 + nr) * HDIM;
    Frag qa0, qa1;
    qa0.q[0] = *(const u32x4*)(qbase + kb);
    qa0.q[1] = *(const u32x4*)(qbase + 16 + kb);
    qa1.q[0] = *(const u32x4*)(qbase + 32 + kb);
    qa1.q[1] = *(const u32x4*)(qbase + 48 + kb);

    const v8f zero = {0.f,0.f,0.f,0.f,0.f,0.f,0.f,0.f};
    v8f o[4] = {zero, zero, zero, zero};
    float m_r[8], l_r[8];
#pragma unroll
    for (int v = 0; v < 8; ++v) { m_r[v] = -1e30f; l_r[v] = 0.f; }

    for (int kt = 0; kt < T_TOK; kt += 32) {
        v8f s0 = zero, s1 = zero;
        {   // scores for keys [kt, kt+16)
            const unsigned short* kp = Kh + ((size_t)h * T_TOK + kt + nr) * HDIM;
            Frag kf;
            kf.q[0] = *(const u32x4*)(kp + kb);
            kf.q[1] = *(const u32x4*)(kp + 16 + kb);
            s0 = __builtin_amdgcn_wmma_f32_16x16x32_bf16(false, qa0.v, false, kf.v, (short)0, s0, false, false);
            kf.q[0] = *(const u32x4*)(kp + 32 + kb);
            kf.q[1] = *(const u32x4*)(kp + 48 + kb);
            s0 = __builtin_amdgcn_wmma_f32_16x16x32_bf16(false, qa1.v, false, kf.v, (short)0, s0, false, false);
        }
        {   // scores for keys [kt+16, kt+32)
            const unsigned short* kp = Kh + ((size_t)h * T_TOK + kt + 16 + nr) * HDIM;
            Frag kf;
            kf.q[0] = *(const u32x4*)(kp + kb);
            kf.q[1] = *(const u32x4*)(kp + 16 + kb);
            s1 = __builtin_amdgcn_wmma_f32_16x16x32_bf16(false, qa0.v, false, kf.v, (short)0, s1, false, false);
            kf.q[0] = *(const u32x4*)(kp + 32 + kb);
            kf.q[1] = *(const u32x4*)(kp + 48 + kb);
            s1 = __builtin_amdgcn_wmma_f32_16x16x32_bf16(false, qa1.v, false, kf.v, (short)0, s1, false, false);
        }

        // Preload V B-fragments (independent of scores) so their latency is
        // covered by the softmax VALU work below. V is [h][dim][t]:
        // lane -> dim column, elements -> contiguous keys => b128 loads.
        Frag vf[4];
#pragma unroll
        for (int j = 0; j < 4; ++j) {
            const unsigned short* vr =
                Vt + (size_t)(h*HDIM + j*16 + nr) * T_TOK + kt + kb;
            vf[j].q[0] = *(const u32x4*)vr;
            vf[j].q[1] = *(const u32x4*)(vr + 16);
        }

        // online softmax update; stage P tile (16x32) into LDS as bf16
        float c_r[8];
#pragma unroll
        for (int v = 0; v < 8; ++v) {
            float a0 = s0[v] * scale, a1 = s1[v] * scale;
            float tm = fmaxf(a0, a1);
            tm = fmaxf(tm, __shfl_xor(tm, 1));
            tm = fmaxf(tm, __shfl_xor(tm, 2));
            tm = fmaxf(tm, __shfl_xor(tm, 4));
            tm = fmaxf(tm, __shfl_xor(tm, 8));
            float nm = fmaxf(m_r[v], tm);
            float p0 = __expf(a0 - nm);
            float p1 = __expf(a1 - nm);
            float ts = p0 + p1;
            ts += __shfl_xor(ts, 1);
            ts += __shfl_xor(ts, 2);
            ts += __shfl_xor(ts, 4);
            ts += __shfl_xor(ts, 8);
            float corr = __expf(m_r[v] - nm);
            l_r[v] = l_r[v] * corr + ts;
            m_r[v] = nm;
            c_r[v] = corr;
            const int prow = v + rofs;
            pst[prow*32 + nr]      = f2bf(p0);
            pst[prow*32 + 16 + nr] = f2bf(p1);
        }
#pragma unroll
        for (int j = 0; j < 4; ++j)
#pragma unroll
            for (int v = 0; v < 8; ++v) o[j][v] *= c_r[v];

        asm volatile("s_wait_dscnt 0" ::: "memory");   // P staging visible in-wave

        // P A-fragment (M=query, K=32 keys)
        Frag pa;
#pragma unroll
        for (int e = 0; e < 8; ++e) {
            pa.u[e]     = pst[nr*32 + kb + e];
            pa.u[8 + e] = pst[nr*32 + 16 + kb + e];
        }

        // P @ V with preloaded fragments
#pragma unroll
        for (int j = 0; j < 4; ++j) {
            o[j] = __builtin_amdgcn_wmma_f32_16x16x32_bf16(
                false, pa.v, false, vf[j].v, (short)0, o[j], false, false);
        }
    }

    // normalize and store (bf16, interleaved back to [t][h*64+d])
#pragma unroll
    for (int j = 0; j < 4; ++j) {
#pragma unroll
        for (int v = 0; v < 8; ++v) {
            const float val = o[j][v] / l_r[v];
            const int row = qt*16 + v + rofs;
            const int col = h*HDIM + j*16 + nr;
            Obf[(size_t)row * MDIM + col] = f2bf(val);
        }
    }
}

// =====================================================================
// Small helper kernels
// =====================================================================
__global__ void cvt_bf16_kernel(const float* __restrict__ x,
                                unsigned short* __restrict__ y, int n) {
    int i = blockIdx.x * blockDim.x + threadIdx.x;
    if (i < n) y[i] = f2bf(x[i]);
}

__global__ void copy_f32_bf16_kernel(const float* __restrict__ x,
                                     float* __restrict__ yf,
                                     unsigned short* __restrict__ yb, int n) {
    int i = blockIdx.x * blockDim.x + threadIdx.x;
    if (i < n) { float v = x[i]; yf[i] = v; yb[i] = f2bf(v); }
}

// qkv (T x 1536 f32) -> Q,K bf16 [h][t][d] ; V bf16 TRANSPOSED [h][d][t]
__global__ void qkv_repack_kernel(const float* __restrict__ qkv,
                                  unsigned short* __restrict__ q,
                                  unsigned short* __restrict__ k,
                                  unsigned short* __restrict__ v) {
    int idx = blockIdx.x * blockDim.x + threadIdx.x;
    if (idx >= T_TOK * MDIM) return;
    int t  = idx / MDIM;
    int c  = idx - t * MDIM;
    int hh = c / HDIM;
    int d  = c - hh * HDIM;
    size_t src = (size_t)t * (3 * MDIM);
    size_t dst = ((size_t)hh * T_TOK + t) * HDIM + d;
    q[dst] = f2bf(qkv[src + c]);
    k[dst] = f2bf(qkv[src + MDIM + c]);
    v[((size_t)hh * HDIM + d) * T_TOK + t] = f2bf(qkv[src + 2*MDIM + c]);
}

// residual + LayerNorm (fused), emits f32 (in-place) and bf16 copies
__global__ void __launch_bounds__(128) ln_residual_kernel(
    float* __restrict__ h, const float* __restrict__ delta,
    const float* __restrict__ g, const float* __restrict__ b,
    unsigned short* __restrict__ hbf)
{
    __shared__ float red[128];
    const int row = blockIdx.x;
    const int tid = threadIdx.x;
    const size_t base = (size_t)row * MDIM;
    float x[4];
    float s = 0.f;
#pragma unroll
    for (int i = 0; i < 4; ++i) {
        int c = tid*4 + i;
        x[i] = h[base + c] + delta[base + c];
        s += x[i];
    }
    red[tid] = s; __syncthreads();
    for (int st = 64; st > 0; st >>= 1) { if (tid < st) red[tid] += red[tid + st]; __syncthreads(); }
    const float mu = red[0] * (1.f / MDIM);
    __syncthreads();
    float vs = 0.f;
#pragma unroll
    for (int i = 0; i < 4; ++i) { float d = x[i] - mu; vs += d * d; }
    red[tid] = vs; __syncthreads();
    for (int st = 64; st > 0; st >>= 1) { if (tid < st) red[tid] += red[tid + st]; __syncthreads(); }
    const float inv = rsqrtf(red[0] * (1.f / MDIM) + 1e-5f);
#pragma unroll
    for (int i = 0; i < 4; ++i) {
        int c = tid*4 + i;
        float y = (x[i] - mu) * inv * g[c] + b[c];
        h[base + c]   = y;
        hbf[base + c] = f2bf(y);
    }
}

// A[r] = sum_c tanh(V[r,c]) * sigmoid(U[r,c]) * Ww[c] + bw   (one wave per row)
__global__ void __launch_bounds__(256) gate_score_kernel(
    const float* __restrict__ Vg, const float* __restrict__ Ug,
    const float* __restrict__ Ww, const float* __restrict__ bw,
    float* __restrict__ A, int R)
{
    const int lane = threadIdx.x & 31;
    const int row  = blockIdx.x * 8 + (threadIdx.x >> 5);
    if (row >= R) return;
    float s = 0.f;
#pragma unroll
    for (int i = 0; i < 4; ++i) {
        int c = lane*4 + i;
        float t  = tanhf(Vg[(size_t)row * LGATE + c]);
        float u  = Ug[(size_t)row * LGATE + c];
        float sg = 1.f / (1.f + __expf(-u));
        s += t * sg * Ww[c];
    }
    s += __shfl_xor(s, 1);
    s += __shfl_xor(s, 2);
    s += __shfl_xor(s, 4);
    s += __shfl_xor(s, 8);
    s += __shfl_xor(s, 16);
    if (lane == 0) A[row] = s + bw[0];
}

// per-segment softmax + weighted pooling (segments are contiguous, <=16 long)
__global__ void __launch_bounds__(256) segpool_kernel(
    const float* __restrict__ A, const float* __restrict__ X,
    float* __restrict__ out, float* __restrict__ wout, int segLen, int ncols)
{
    __shared__ float w[16];
    const int seg = blockIdx.x;
    const int tid = threadIdx.x;
    if (tid == 0) {
        float mx = -1e30f;
        for (int i = 0; i < segLen; ++i) mx = fmaxf(mx, A[seg*segLen + i]);
        float sum = 0.f;
        for (int i = 0; i < segLen; ++i) { float e = __expf(A[seg*segLen + i] - mx); w[i] = e; sum += e; }
        float inv = 1.f / sum;
        for (int i = 0; i < segLen; ++i) { w[i] *= inv; wout[seg*segLen + i] = w[i]; }
    }
    __syncthreads();
    for (int c = tid; c < ncols; c += blockDim.x) {
        float acc = 0.f;
        for (int i = 0; i < segLen; ++i)
            acc += w[i] * X[(size_t)(seg*segLen + i) * ncols + c];
        out[(size_t)seg * ncols + c] = acc;
    }
}

// conv1d + bias + BN(eval) + ReLU (same-length output via asymmetric pad)
__global__ void conv_bn_relu_kernel(
    const float* __restrict__ x, int Cin, int Lin,
    const float* __restrict__ Kc, int ksz, int padL,
    const float* __restrict__ bc, const float* __restrict__ g, const float* __restrict__ bb,
    float* __restrict__ y, int Cout, int Nb)
{
    int idx = blockIdx.x * blockDim.x + threadIdx.x;
    int total = Nb * Cout * Lin;
    if (idx >= total) return;
    int p   = idx % Lin;
    int tmp = idx / Lin;
    int co  = tmp % Cout;
    int n   = tmp / Cout;
    float acc = bc[co];
    for (int ci = 0; ci < Cin; ++ci) {
        const float* xr = x + ((size_t)n * Cin + ci) * Lin;
        const float* kr = Kc + ((size_t)co * Cin + ci) * ksz;
        for (int kk = 0; kk < ksz; ++kk) {
            int qq = p + kk - padL;
            if (qq >= 0 && qq < Lin) acc += xr[qq] * kr[kk];
        }
    }
    float scale = g[co] * rsqrtf(1.f + 1e-5f);
    y[idx] = fmaxf(acc * scale + bb[co], 0.f);
}

__global__ void pool2_kernel(const float* __restrict__ x, float* __restrict__ y, int n) {
    int i = blockIdx.x * blockDim.x + threadIdx.x;
    if (i < n) y[i] = 0.5f * (x[2*i] + x[2*i + 1]);
}

// FC + ReLU + BN(eval)
__global__ void fc_relu_bn_kernel(
    const float* __restrict__ x, const float* __restrict__ W, const float* __restrict__ b,
    const float* __restrict__ g, const float* __restrict__ bb,
    float* __restrict__ y, int R, int Cin, int Cout)
{
    int idx = blockIdx.x * blockDim.x + threadIdx.x;
    if (idx >= R * Cout) return;
    int o = idx % Cout;
    int r = idx / Cout;
    float acc = b[o];
    const float* xr = x + (size_t)r * Cin;
    const float* wr = W + (size_t)o * Cin;
    for (int c = 0; c < Cin; ++c) acc += xr[c] * wr[c];
    acc = fmaxf(acc, 0.f);
    float scale = g[o] * rsqrtf(1.f + 1e-5f);
    y[idx] = acc * scale + bb[o];
}

__global__ void final_kernel(const float* __restrict__ x, const float* __restrict__ W,
                             const float* __restrict__ b, float* __restrict__ out) {
    int n = threadIdx.x;
    if (n >= NOUTER) return;
    float acc = b[0];
    for (int c = 0; c < 128; ++c) acc += x[n*128 + c] * W[c];
    float p = 1.f / (1.f + __expf(-acc));
    out[n] = p;                                   // Y_prob
    out[NOUTER + n] = (p >= 0.5f) ? 1.f : 0.f;    // Y_hat
}

// =====================================================================
// Host launcher
// =====================================================================
extern "C" void kernel_launch(void* const* d_in, const int* in_sizes, int n_in,
                              void* d_out, int out_size, void* d_ws, size_t ws_size,
                              hipStream_t stream)
{
    (void)in_sizes; (void)n_in; (void)out_size; (void)ws_size;

    const float* datas = (const float*)d_in[0];
    const float* Wqkv  = (const float*)d_in[6];
    const float* bqkv  = (const float*)d_in[7];
    const float* Wo    = (const float*)d_in[8];
    const float* bo    = (const float*)d_in[9];
    const float* ln1g  = (const float*)d_in[10];
    const float* ln1b  = (const float*)d_in[11];
    const float* W1    = (const float*)d_in[12];
    const float* b1    = (const float*)d_in[13];
    const float* W2    = (const float*)d_in[14];
    const float* b2    = (const float*)d_in[15];
    const float* ln2g  = (const float*)d_in[16];
    const float* ln2b  = (const float*)d_in[17];
    const float* Wv1   = (const float*)d_in[18];
    const float* bv1   = (const float*)d_in[19];
    const float* Wu1   = (const float*)d_in[20];
    const float* bu1   = (const float*)d_in[21];
    const float* Ww1   = (const float*)d_in[22];
    const float* bw1   = (const float*)d_in[23];
    const float* Wv2   = (const float*)d_in[24];
    const float* bv2   = (const float*)d_in[25];
    const float* Wu2   = (const float*)d_in[26];
    const float* bu2   = (const float*)d_in[27];
    const float* Ww2   = (const float*)d_in[28];
    const float* bw2   = (const float*)d_in[29];
    const float* Kc1   = (const float*)d_in[30];
    const float* bc1   = (const float*)d_in[31];
    const float* gc1   = (const float*)d_in[32];
    const float* bb1   = (const float*)d_in[33];
    const float* Kc2   = (const float*)d_in[34];
    const float* bc2   = (const float*)d_in[35];
    const float* gc2   = (const float*)d_in[36];
    const float* bb2   = (const float*)d_in[37];
    const float* Kc3   = (const float*)d_in[38];
    const float* bc3   = (const float*)d_in[39];
    const float* gc3   = (const float*)d_in[40];
    const float* bb3   = (const float*)d_in[41];
    const float* Wf1   = (const float*)d_in[42];
    const float* bf1   = (const float*)d_in[43];
    const float* g4    = (const float*)d_in[44];
    const float* b4    = (const float*)d_in[45];
    const float* Wf2   = (const float*)d_in[46];
    const float* bf2   = (const float*)d_in[47];
    const float* g5    = (const float*)d_in[48];
    const float* b5    = (const float*)d_in[49];
    const float* Wf3   = (const float*)d_in[50];
    const float* bf3   = (const float*)d_in[51];

    // -------- workspace layout (~97 MB) --------
    const size_t OFF_H   = 0;                                    // h        f32  (8 MB)
    const size_t OFF_HBF = OFF_H   + (size_t)T_TOK*MDIM*4;       // h        bf16 (4 MB)
    const size_t OFF_W   = OFF_HBF + (size_t)T_TOK*MDIM*2;       // weights  bf16 (12.5 MB)
    const size_t WELEMS  = 6553600;
    const size_t OFF_A   = OFF_W   + WELEMS*2;                   // big scratch f32 (32 MB)
    const size_t OFF_ABF = OFF_A   + (size_t)T_TOK*FFDIM*4;      // big scratch bf16 (16 MB)
    const size_t OFF_B   = OFF_ABF + (size_t)T_TOK*FFDIM*2;      // attn-out  (8 MB)
    const size_t OFF_C   = OFF_B   + (size_t)T_TOK*MDIM*4;       // proj/ff2  (8 MB)
    const size_t OFF_S   = OFF_C   + (size_t)T_TOK*MDIM*4;       // small region (4 MB)

    char* ws = (char*)d_ws;
    float*          H    = (float*)(ws + OFF_H);
    unsigned short* HBF  = (unsigned short*)(ws + OFF_HBF);
    unsigned short* WB   = (unsigned short*)(ws + OFF_W);
    float*          BA   = (float*)(ws + OFF_A);
    unsigned short* BABF = (unsigned short*)(ws + OFF_ABF);
    float*          BB   = (float*)(ws + OFF_B);
    float*          BC   = (float*)(ws + OFF_C);
    float*          SS   = (float*)(ws + OFF_S);
    float*          outp = (float*)d_out;

    // bf16 weight slices
    unsigned short* WQKVb = WB;                 // 2 x 1536 x 512
    unsigned short* WOb   = WB + 1572864;       // 2 x  512 x 512
    unsigned short* W1b   = WB + 2097152;       // 2 x 2048 x 512
    unsigned short* W2b   = WB + 4194304;       // 2 x  512 x 2048
    unsigned short* WV1b  = WB + 6291456;       // 128 x 512
    unsigned short* WU1b  = WB + 6356992;
    unsigned short* WV2b  = WB + 6422528;
    unsigned short* WU2b  = WB + 6488064;

    auto cvt = [&](const float* src, unsigned short* dst, int n) {
        cvt_bf16_kernel<<<(n + 255) / 256, 256, 0, stream>>>(src, dst, n);
    };
    cvt(Wqkv, WQKVb, 2 * 3 * MDIM * MDIM);
    cvt(Wo,   WOb,   2 * MDIM * MDIM);
    cvt(W1,   W1b,   2 * FFDIM * MDIM);
    cvt(W2,   W2b,   2 * MDIM * FFDIM);
    cvt(Wv1,  WV1b,  LGATE * MDIM);
    cvt(Wu1,  WU1b,  LGATE * MDIM);
    cvt(Wv2,  WV2b,  LGATE * MDIM);
    cvt(Wu2,  WU2b,  LGATE * MDIM);

    copy_f32_bf16_kernel<<<(T_TOK*MDIM + 255)/256, 256, 0, stream>>>(datas, H, HBF, T_TOK*MDIM);

    auto gemm = [&](const unsigned short* A_, const unsigned short* B_, const float* bias_,
                    float* C_, unsigned short* Cbf_, int M_, int N_, int K_, int act_) {
        int waves = (M_ / 16) * (N_ / 64);
        gemm_bf16_kernel<<<(waves + 7) / 8, 256, 0, stream>>>(A_, B_, bias_, C_, Cbf_, M_, N_, K_, act_);
    };

    unsigned short* Qb = BABF;
    unsigned short* Kb = BABF + 2097152;
    unsigned short* Vb = BABF + 4194304;   // transposed [h][d][t]

    // -------- transformer layers --------
    for (int l = 0; l < NLAYERS; ++l) {
        gemm(HBF, WQKVb + (size_t)l*786432, bqkv + l*1536, BA, nullptr, T_TOK, 1536, 512, 0);
        qkv_repack_kernel<<<(T_TOK*MDIM + 255)/256, 256, 0, stream>>>(BA, Qb, Kb, Vb);
        attn_kernel<<<(NHEADS * (T_TOK/16)) / 8, 256, 0, stream>>>(Qb, Kb, Vb, (unsigned short*)BB);
        gemm((unsigned short*)BB, WOb + (size_t)l*262144, bo + l*512, BC, nullptr, T_TOK, 512, 512, 0);
        ln_residual_kernel<<<T_TOK, 128, 0, stream>>>(H, BC, ln1g + l*512, ln1b + l*512, HBF);
        gemm(HBF, W1b + (size_t)l*1048576, b1 + l*2048, BA, BABF, T_TOK, 2048, 512, 1);
        gemm(BABF, W2b + (size_t)l*1048576, b2 + l*512, BC, nullptr, T_TOK, 512, 2048, 0);
        ln_residual_kernel<<<T_TOK, 128, 0, stream>>>(H, BC, ln2g + l*512, ln2b + l*512, HBF);
    }

    // -------- gated pooling level 1 --------
    float* Vg = BA;                  // 4096 x 128
    float* Ug = BA + 524288;
    gemm(HBF, WV1b, bv1, Vg, nullptr, T_TOK, 128, 512, 0);
    gemm(HBF, WU1b, bu1, Ug, nullptr, T_TOK, 128, 512, 0);
    float* A1 = SS + 147456;
    gate_score_kernel<<<(T_TOK + 7)/8, 256, 0, stream>>>(Vg, Ug, Ww1, bw1, A1, T_TOK);
    float* out1 = SS;                                            // 256 x 512
    segpool_kernel<<<NINNER, 256, 0, stream>>>(A1, H, out1, outp + 64, 16, 512);

    // -------- gated pooling level 2 --------
    unsigned short* out1bf = (unsigned short*)(SS + 229632);
    cvt(out1, out1bf, NINNER * 512);
    float* V2g = SS + 151808;
    float* U2g = SS + 184576;
    gemm(out1bf, WV2b, bv2, V2g, nullptr, NINNER, 128, 512, 0);
    gemm(out1bf, WU2b, bu2, U2g, nullptr, NINNER, 128, 512, 0);
    float* A2 = SS + 151552;
    gate_score_kernel<<<(NINNER + 7)/8, 256, 0, stream>>>(V2g, U2g, Ww2, bw2, A2, NINNER);
    float* out2 = SS + 131072;                                   // 32 x 512
    segpool_kernel<<<NOUTER, 256, 0, stream>>>(A2, out1, out2, outp + 64 + 4096, 8, 512);

    // -------- CNN / MLP head --------
    float* hc1 = BA;
    float* hc2 = BA + 2097152;
    float* hp2 = BA + 4194304;
    float* hc3 = BA + 5242880;
    float* hp3 = BA + 6291456;
    conv_bn_relu_kernel<<<(32*128*512 + 255)/256, 256, 0, stream>>>(out2, 1,   512, Kc1, 4, 1, bc1, gc1, bb1, hc1, 128, 32);
    conv_bn_relu_kernel<<<(32*128*512 + 255)/256, 256, 0, stream>>>(hc1,  128, 512, Kc2, 5, 2, bc2, gc2, bb2, hc2, 128, 32);
    pool2_kernel<<<(32*128*256 + 255)/256, 256, 0, stream>>>(hc2, hp2, 32*128*256);
    conv_bn_relu_kernel<<<(32*128*256 + 255)/256, 256, 0, stream>>>(hp2,  128, 256, Kc3, 7, 3, bc3, gc3, bb3, hc3, 128, 32);
    pool2_kernel<<<(32*128*128 + 255)/256, 256, 0, stream>>>(hc3, hp3, 32*128*128);

    float* f1 = SS + 217344;   // 32 x 256
    float* f2 = SS + 225536;   // 32 x 128
    fc_relu_bn_kernel<<<(32*256 + 255)/256, 256, 0, stream>>>(hp3, Wf1, bf1, g4, b4, f1, 32, 16384, 256);
    fc_relu_bn_kernel<<<(32*128 + 255)/256, 256, 0, stream>>>(f1,  Wf2, bf2, g5, b5, f2, 32, 256,   128);
    final_kernel<<<1, 32, 0, stream>>>(f2, Wf3, bf3, outp);
}
